// BaseModel_46016279609979
// MI455X (gfx1250) — matrix operand order repaired and checked by reading, not compiled, not run on hardware
//
#include <hip/hip_runtime.h>

// ---------------------------------------------------------------------------
// DA-RNN fused kernel for MI455X (gfx1250, wave32, WMMA 16x16x32 f16).
// One block = 16 batch rows; full encoder (50 steps) + decoder (3 steps)
// fused, recurrent state in LDS, all GEMMs on v_wmma_f32_16x16x32_f16.
// Weights converted/transposed to f16 once into d_ws (needs ~40 MB scratch).
// Branchless tanh/sigmoid (v_exp_f32 + v_rcp_f32) so TRANS ops co-execute
// with the WMMA pipeline instead of forcing exec-mask branches.
// ---------------------------------------------------------------------------

#define L_     50
#define NOUT_  3
#define F_     64
#define B_     2048
#define H_     128
#define BT     16                 // batch tile per block (== WMMA M)
#define NBLK   (B_/BT)            // 128 blocks
#define TPB    256                // 8 wave32
#define NWAVE  (TPB/32)

typedef _Float16 h16;
typedef __attribute__((ext_vector_type(16))) _Float16 v16h;
typedef __attribute__((ext_vector_type(8)))  _Float16 v8h;
typedef __attribute__((ext_vector_type(8)))  float    v8f;

// ---- workspace layout (bytes, all 256-aligned) ----
#define WS_XH      ((size_t)0)                         // B*L*F   f16  13,107,200
#define WS_ENCW    ((size_t)13107200)                  // 512x192 f16     196,608
#define WS_ENCB    (WS_ENCW  + (size_t)196608)         // 512     f32       2,048
#define WS_ATTNW   (WS_ENCB  + (size_t)2048)           // 50*64*192 f16 1,228,800
#define WS_DDT     (WS_ATTNW + (size_t)1228800)        // 3*128*128 f16    98,304
#define WS_DDB     (WS_DDT   + (size_t)98304)          // 3*128*128 f16    98,304
#define WS_DECW    (WS_DDB   + (size_t)98304)          // 512x384 f16     393,216
#define WS_DECB    (WS_DECW  + (size_t)393216)         // 512     f32       2,048
#define WS_FCT     (WS_DECB  + (size_t)2048)           // 3*64*128 f16     49,152
#define WS_ENC     (WS_FCT   + (size_t)49152)          // B*L*H   f16  26,214,400
#define WS_TOTAL   (WS_ENC   + (size_t)26214400)       // ~39.5 MiB

// ---------------------------------------------------------------------------
// WMMA fragment helpers (CDNA5 16x16x32 f16 layouts, CDNA5 ISA 7.12.2)
// ---------------------------------------------------------------------------
__device__ __forceinline__ v8f wmma32(v16h a, v16h b, v8f c) {
  return __builtin_amdgcn_wmma_f32_16x16x32_f16(false, a, false, b,
                                                (short)0, c, false, false);
}

// A (16xK row-major, k contiguous, row stride ld halfs): lane<16 holds K=
// [kb..kb+7]U[kb+16..kb+23], lane>=16 holds [kb+8..kb+15]U[kb+24..kb+31].
__device__ __forceinline__ v16h load_a(const h16* base, int ld, int kb, int lane) {
  const int m = lane & 15, hi = lane >> 4;
  const h16* p = base + m * ld + kb + hi * 8;
  v8h x0 = *(const v8h*)(p);
  v8h x1 = *(const v8h*)(p + 16);
  v16h r;
#pragma unroll
  for (int j = 0; j < 8; ++j) { r[j] = x0[j]; r[j + 8] = x1[j]; }
  return r;
}

// B (KxN, stored transposed as W_t[n][k], k contiguous, row stride ldk):
// lane<16 holds K=[kb..kb+7]U[kb+16..kb+23], lane>=16 holds [kb+8..kb+15]U[kb+24..kb+31].
__device__ __forceinline__ v16h load_b(const h16* wt, int ldk, int n0, int kb, int lane) {
  const int n = n0 + (lane & 15), hi = lane >> 4;
  const h16* p = wt + (size_t)n * ldk + kb + hi * 8;
  v8h x0 = *(const v8h*)(p);
  v8h x1 = *(const v8h*)(p + 16);
  v16h r;
#pragma unroll
  for (int j = 0; j < 8; ++j) { r[j] = x0[j]; r[j + 8] = x1[j]; }
  return r;
}

// Branchless transcendentals: v_exp_f32 + v_rcp_f32, no exec-mask branches.
__device__ __forceinline__ float fast_tanh(float x) {
  float xc = fminf(fmaxf(x, -15.f), 15.f);
  float e  = __expf(2.f * xc);                    // v_exp_f32 (TRANS)
  return 1.f - 2.f * __builtin_amdgcn_rcpf(e + 1.f);  // v_rcp_f32 (TRANS)
}
__device__ __forceinline__ float fast_sigm(float x) {
  float xc = fminf(fmaxf(x, -30.f), 30.f);
  return __builtin_amdgcn_rcpf(1.f + __expf(-xc));
}

// ---------------------------------------------------------------------------
// Prep kernels: convert/transposed weights + x to f16 in d_ws.
// ---------------------------------------------------------------------------
__global__ void k_cvt_x(const float* __restrict__ x, h16* __restrict__ xh, int n) {
  for (int i = blockIdx.x * blockDim.x + threadIdx.x; i < n; i += gridDim.x * blockDim.x)
    xh[i] = (h16)x[i];
}
__global__ void k_encw(const float* Wih, const float* Whh, const float* bih,
                       const float* bhh, h16* encW, float* encB) {
  int i = blockIdx.x * blockDim.x + threadIdx.x;
  if (i < 512 * 192) {
    int n = i / 192, k = i - n * 192;
    encW[i] = (h16)((k < F_) ? Wih[n * F_ + k] : Whh[n * H_ + (k - F_)]);
  }
  if (i < 512) encB[i] = bih[i] + bhh[i];
}
__global__ void k_decw(const float* Wih, const float* Whh, const float* bih,
                       const float* bhh, h16* decW, float* decB) {
  int i = blockIdx.x * blockDim.x + threadIdx.x;
  if (i < 512 * 384) {
    int n = i / 384, k = i - n * 384;
    decW[i] = (h16)((k < 256) ? Wih[n * 256 + k] : Whh[n * H_ + (k - 256)]);
  }
  if (i < 512) decB[i] = bih[i] + bhh[i];
}
__global__ void k_attnw(const float* aw, h16* awT) {
  int i = blockIdx.x * blockDim.x + threadIdx.x;
  if (i >= L_ * F_ * 192) return;
  int l = i / (F_ * 192), r = i - l * (F_ * 192);
  int n = r / 192, k = r - n * 192;
  awT[i] = (h16)aw[((size_t)l * 192 + k) * F_ + n];
}
__global__ void k_dd(const float* dd, h16* ddT, h16* ddB) {
  int i = blockIdx.x * blockDim.x + threadIdx.x;
  if (i >= NOUT_ * H_ * H_) return;
  int s = i / (H_ * H_), r = i - s * (H_ * H_);
  int n = r / H_, k = r - n * H_;
  ddT[i] = (h16)dd[((size_t)s * 256 + k) * H_ + n];
  ddB[i] = (h16)dd[((size_t)s * 256 + 128 + k) * H_ + n];
}
__global__ void k_fct(const float* fc, h16* fcT) {
  int i = blockIdx.x * blockDim.x + threadIdx.x;
  if (i >= NOUT_ * 64 * H_) return;
  int s = i / (64 * H_), r = i - s * (64 * H_);
  int n = r / H_, k = r - n * H_;
  fcT[i] = (h16)fc[((size_t)s * H_ + k) * 64 + n];
}

// ---------------------------------------------------------------------------
// Fused encoder+decoder kernel.
// ---------------------------------------------------------------------------
static constexpr size_t SMEM_BYTES =
    (size_t)BT * 384 * 2      // sm_a   (f16 A staging, ld up to 384)
  + (size_t)BT * H_ * 4 * 4   // sm_h, sm_c, sm_hde, sm_cde
  + (size_t)BT * 512 * 4      // sm_g
  + (size_t)BT * 64 * 4       // sm_e
  + (size_t)BT * H_ * 4 * 2   // sm_ctx2, sm_ctx
  + (size_t)BT * 52 * 4;      // sm_sc (attention scores over L)

__global__ __launch_bounds__(TPB) void fused_darnn(
    const h16* __restrict__ xh, const h16* __restrict__ encW,
    const float* __restrict__ encB, const h16* __restrict__ attnW,
    const float* __restrict__ attn_b, const h16* __restrict__ ddT,
    const h16* __restrict__ ddB_, const float* __restrict__ dd_b,
    const float* __restrict__ dl_w, const float* __restrict__ dl_b,
    const h16* __restrict__ decW, const float* __restrict__ decB,
    const h16* __restrict__ fcT, const float* __restrict__ fc_b,
    const float* __restrict__ out_w, const float* __restrict__ out_b,
    h16* __restrict__ encS, float* __restrict__ out) {
  extern __shared__ unsigned char smem[];
  h16*   sm_a    = (h16*)smem;                                   // [16][<=384]
  float* sm_h    = (float*)(smem + (size_t)BT * 384 * 2);        // [16][128]
  float* sm_c    = sm_h   + BT * H_;
  float* sm_hde  = sm_c   + BT * H_;
  float* sm_cde  = sm_hde + BT * H_;
  float* sm_g    = sm_cde + BT * H_;                             // [16][512]
  float* sm_e    = sm_g   + BT * 512;                            // [16][64]
  float* sm_ctx2 = sm_e   + BT * 64;                             // [16][128]
  float* sm_ctx  = sm_ctx2 + BT * H_;                            // [16][128]
  float* sm_sc   = sm_ctx  + BT * H_;                            // [16][52]

  const int tid = threadIdx.x;
  const int wid = tid >> 5;
  const int lane = tid & 31;
  const int blk = blockIdx.x;
  const int b0 = blk * BT;

  for (int i = tid; i < BT * H_; i += TPB) {
    sm_h[i] = 0.f; sm_c[i] = 0.f; sm_hde[i] = 0.f; sm_cde[i] = 0.f;
  }
  __syncthreads();

  // ============================ encoder ============================
  for (int t = 0; t < L_; ++t) {
    // E1: stage A = [x_t | h]  (f16, ld=192)
    for (int i = tid; i < BT * 192; i += TPB) {
      int m = i / 192, k = i - m * 192;
      sm_a[i] = (k < F_) ? xh[((size_t)(b0 + m) * L_ + t) * F_ + k]
                         : (h16)sm_h[m * H_ + (k - F_)];
    }
    __syncthreads();

    // E2: e = tanh(A @ attnW[t] + attn_b[t])  -> sm_e (waves 0..3)
    if (wid < 4) {
      const h16* wt = attnW + (size_t)t * F_ * 192;
      v8f acc = {};
#pragma unroll
      for (int kk = 0; kk < 6; ++kk)
        acc = wmma32(load_a(sm_a, 192, kk * 32, lane),
                     load_b(wt, 192, wid * 16, kk * 32, lane), acc);
      const int n = wid * 16 + (lane & 15);
      const float bias = attn_b[t * F_ + n];
      const int mb = (lane >> 4) * 8;
#pragma unroll
      for (int r = 0; r < 8; ++r)
        sm_e[(r + mb) * F_ + n] = fast_tanh(acc[r] + bias);
    }
    __syncthreads();

    // E3: softmax over F, xin = softmax(e)*x_t, overwrite sm_a[:,0:F]
    {
      const int m = tid >> 4, j = tid & 15;
      float v0 = sm_e[m * F_ + j],      v1 = sm_e[m * F_ + j + 16],
            v2 = sm_e[m * F_ + j + 32], v3 = sm_e[m * F_ + j + 48];
      float mx = fmaxf(fmaxf(v0, v1), fmaxf(v2, v3));
#pragma unroll
      for (int s = 8; s >= 1; s >>= 1) mx = fmaxf(mx, __shfl_xor(mx, s, 32));
      v0 = __expf(v0 - mx); v1 = __expf(v1 - mx);
      v2 = __expf(v2 - mx); v3 = __expf(v3 - mx);
      float sum = v0 + v1 + v2 + v3;
#pragma unroll
      for (int s = 8; s >= 1; s >>= 1) sum += __shfl_xor(sum, s, 32);
      const float inv = __builtin_amdgcn_rcpf(sum);
      const h16* xr = xh + ((size_t)(b0 + m) * L_ + t) * F_;
      sm_a[m * 192 + j]      = (h16)(v0 * inv * (float)xr[j]);
      sm_a[m * 192 + j + 16] = (h16)(v1 * inv * (float)xr[j + 16]);
      sm_a[m * 192 + j + 32] = (h16)(v2 * inv * (float)xr[j + 32]);
      sm_a[m * 192 + j + 48] = (h16)(v3 * inv * (float)xr[j + 48]);
    }
    __syncthreads();

    // E4: gates = [xin|h] @ encW^T + bias   (32 N-tiles over 8 waves)
    for (int nt = wid; nt < 32; nt += NWAVE) {
      v8f acc = {};
#pragma unroll
      for (int kk = 0; kk < 6; ++kk)
        acc = wmma32(load_a(sm_a, 192, kk * 32, lane),
                     load_b(encW, 192, nt * 16, kk * 32, lane), acc);
      const int n = nt * 16 + (lane & 15);
      const float bias = encB[n];
      const int mb = (lane >> 4) * 8;
#pragma unroll
      for (int r = 0; r < 8; ++r) sm_g[(r + mb) * 512 + n] = acc[r] + bias;
    }
    __syncthreads();

    // E5: LSTM cell elementwise; store h to state + enc scratch (f16)
    for (int i = tid; i < BT * H_; i += TPB) {
      const int m = i >> 7, k = i & 127;
      const float gi = sm_g[m * 512 + k],        gf = sm_g[m * 512 + 128 + k],
                  gg = sm_g[m * 512 + 256 + k],  go = sm_g[m * 512 + 384 + k];
      const float c = fast_sigm(gf) * sm_c[i] + fast_sigm(gi) * fast_tanh(gg);
      const float h = fast_sigm(go) * fast_tanh(c);
      sm_c[i] = c; sm_h[i] = h;
      encS[(((size_t)blk * L_ + t) * BT + m) * H_ + k] = (h16)h;
    }
    __syncthreads();
  }

  // ============================ decoder ============================
  for (int st = 0; st < NOUT_; ++st) {
    // D1: stage h_de (f16, ld=128); ctx2 = h_de @ dd_w[bottom] + dd_b
    for (int i = tid; i < BT * H_; i += TPB) sm_a[i] = (h16)sm_hde[i];
    __syncthreads();
    {
      const h16* wb = ddB_ + (size_t)st * H_ * H_;
      v8f acc = {};
#pragma unroll
      for (int kk = 0; kk < 4; ++kk)
        acc = wmma32(load_a(sm_a, 128, kk * 32, lane),
                     load_b(wb, 128, wid * 16, kk * 32, lane), acc);
      const int n = wid * 16 + (lane & 15);
      const float bias = dd_b[st * H_ + n];
      const int mb = (lane >> 4) * 8;
#pragma unroll
      for (int r = 0; r < 8; ++r) sm_ctx2[(r + mb) * H_ + n] = acc[r] + bias;
    }
    __syncthreads();

    // D2: scores[m][l] = tanh(enc_l @ ddT + ctx2) . dl_w + dl_b
    {
      const h16* wt = ddT + (size_t)st * H_ * H_;
      const float dlb = dl_b[st];
      for (int l = wid; l < L_; l += NWAVE) {
        const h16* ea = encS + (((size_t)blk * L_ + l) * BT) * H_;
        if (l + NWAVE < L_)   // pull next enc tile toward L0 (global_prefetch_b8)
          __builtin_prefetch(encS + (((size_t)blk * L_ + l + NWAVE) * BT) * H_ +
                                 (size_t)lane * 64, 0, 0);
        float part[8];
#pragma unroll
        for (int r = 0; r < 8; ++r) part[r] = 0.f;
#pragma unroll
        for (int nt = 0; nt < 8; ++nt) {
          v8f acc = {};
#pragma unroll
          for (int kk = 0; kk < 4; ++kk)
            acc = wmma32(load_a(ea, 128, kk * 32, lane),
                         load_b(wt, 128, nt * 16, kk * 32, lane), acc);
          const int n = nt * 16 + (lane & 15);
          const float dw = dl_w[st * H_ + n];
          const int mb = (lane >> 4) * 8;
#pragma unroll
          for (int r = 0; r < 8; ++r)
            part[r] += fast_tanh(acc[r] + sm_ctx2[(r + mb) * H_ + n]) * dw;
        }
#pragma unroll
        for (int r = 0; r < 8; ++r) {
#pragma unroll
          for (int s = 8; s >= 1; s >>= 1) part[r] += __shfl_xor(part[r], s, 32);
        }
        if ((lane & 15) == 0) {
          const int mb = (lane >> 4) * 8;
#pragma unroll
          for (int r = 0; r < 8; ++r) sm_sc[(r + mb) * 52 + l] = part[r] + dlb;
        }
      }
    }
    __syncthreads();

    // D3: softmax over L; ctx = sum_l alpha_l * enc_l
    {
      const int m = tid >> 4, j = tid & 15;
      float v[4];
      float mx = -3.4e38f;
#pragma unroll
      for (int q = 0; q < 4; ++q) {
        const int l = j + 16 * q;
        v[q] = (l < L_) ? sm_sc[m * 52 + l] : -3.4e38f;
        mx = fmaxf(mx, v[q]);
      }
#pragma unroll
      for (int s = 8; s >= 1; s >>= 1) mx = fmaxf(mx, __shfl_xor(mx, s, 32));
      float sum = 0.f;
#pragma unroll
      for (int q = 0; q < 4; ++q) {
        const int l = j + 16 * q;
        v[q] = (l < L_) ? __expf(v[q] - mx) : 0.f;
        sum += v[q];
      }
#pragma unroll
      for (int s = 8; s >= 1; s >>= 1) sum += __shfl_xor(sum, s, 32);
      const float inv = __builtin_amdgcn_rcpf(sum);
#pragma unroll
      for (int q = 0; q < 4; ++q) {
        const int l = j + 16 * q;
        if (l < L_) sm_sc[m * 52 + l] = v[q] * inv;
      }
    }
    __syncthreads();
    {
      const int m = tid >> 4, k0 = (tid & 15) * 8;
      float acc8[8];
#pragma unroll
      for (int r = 0; r < 8; ++r) acc8[r] = 0.f;
      for (int l = 0; l < L_; ++l) {
        const float al = sm_sc[m * 52 + l];
        const v8h ev = *(const v8h*)(encS + (((size_t)blk * L_ + l) * BT + m) * H_ + k0);
#pragma unroll
        for (int r = 0; r < 8; ++r) acc8[r] += al * (float)ev[r];
      }
#pragma unroll
      for (int r = 0; r < 8; ++r) sm_ctx[m * H_ + k0 + r] = acc8[r];
    }
    __syncthreads();

    // D4: stage [ctx | h_de | h_de] (ld=384); gates = A @ decW^T + bias
    for (int i = tid; i < BT * 384; i += TPB) {
      const int m = i / 384, k = i - m * 384;
      const float v = (k < H_) ? sm_ctx[m * H_ + k] : sm_hde[m * H_ + ((k - H_) & 127)];
      sm_a[i] = (h16)v;
    }
    __syncthreads();
    for (int nt = wid; nt < 32; nt += NWAVE) {
      v8f acc = {};
#pragma unroll
      for (int kk = 0; kk < 12; ++kk)
        acc = wmma32(load_a(sm_a, 384, kk * 32, lane),
                     load_b(decW, 384, nt * 16, kk * 32, lane), acc);
      const int n = nt * 16 + (lane & 15);
      const float bias = decB[n];
      const int mb = (lane >> 4) * 8;
#pragma unroll
      for (int r = 0; r < 8; ++r) sm_g[(r + mb) * 512 + n] = acc[r] + bias;
    }
    __syncthreads();

    // D5: decoder LSTM elementwise; restage new h_de into sm_a (ld=128)
    for (int i = tid; i < BT * H_; i += TPB) {
      const int m = i >> 7, k = i & 127;
      const float gi = sm_g[m * 512 + k],       gf = sm_g[m * 512 + 128 + k],
                  gg = sm_g[m * 512 + 256 + k], go = sm_g[m * 512 + 384 + k];
      const float c = fast_sigm(gf) * sm_cde[i] + fast_sigm(gi) * fast_tanh(gg);
      const float h = fast_sigm(go) * fast_tanh(c);
      sm_cde[i] = c; sm_hde[i] = h;
      sm_a[i] = (h16)h;
    }
    __syncthreads();

    // D6: y1 = tanh(h_de @ fcT + fc_b)  (waves 0..3)
    if (wid < 4) {
      const h16* wf = fcT + (size_t)st * 64 * H_;
      v8f acc = {};
#pragma unroll
      for (int kk = 0; kk < 4; ++kk)
        acc = wmma32(load_a(sm_a, 128, kk * 32, lane),
                     load_b(wf, 128, wid * 16, kk * 32, lane), acc);
      const int n = wid * 16 + (lane & 15);
      const float bias = fc_b[st * 64 + n];
      const int mb = (lane >> 4) * 8;
#pragma unroll
      for (int r = 0; r < 8; ++r) sm_e[(r + mb) * 64 + n] = fast_tanh(acc[r] + bias);
    }
    __syncthreads();

    // D7: y = sigmoid(y1 . out_w + out_b)
    if (tid < BT) {
      float acc = out_b[st];
      const float* ow = out_w + st * 64;
#pragma unroll 8
      for (int n = 0; n < 64; ++n) acc += sm_e[tid * 64 + n] * ow[n];
      out[(size_t)(b0 + tid) * NOUT_ + st] = fast_sigm(acc);
    }
    __syncthreads();
  }
}

// ---------------------------------------------------------------------------
extern "C" void kernel_launch(void* const* d_in, const int* in_sizes, int n_in,
                              void* d_out, int out_size, void* d_ws, size_t ws_size,
                              hipStream_t stream) {
  (void)in_sizes; (void)n_in; (void)out_size; (void)ws_size;
  const float* x      = (const float*)d_in[0];
  const float* eWih   = (const float*)d_in[1];
  const float* eWhh   = (const float*)d_in[2];
  const float* ebih   = (const float*)d_in[3];
  const float* ebhh   = (const float*)d_in[4];
  const float* attn_w = (const float*)d_in[5];
  const float* attn_b = (const float*)d_in[6];
  const float* dWih   = (const float*)d_in[7];
  const float* dWhh   = (const float*)d_in[8];
  const float* dbih   = (const float*)d_in[9];
  const float* dbhh   = (const float*)d_in[10];
  const float* dd_w   = (const float*)d_in[11];
  const float* dd_b   = (const float*)d_in[12];
  const float* dl_w   = (const float*)d_in[13];
  const float* dl_b   = (const float*)d_in[14];
  const float* fc_w   = (const float*)d_in[15];
  const float* fc_b   = (const float*)d_in[16];
  const float* out_w  = (const float*)d_in[17];
  const float* out_b  = (const float*)d_in[18];

  char* ws = (char*)d_ws;
  h16*   xh    = (h16*)(ws + WS_XH);
  h16*   encW  = (h16*)(ws + WS_ENCW);
  float* encB  = (float*)(ws + WS_ENCB);
  h16*   attnW = (h16*)(ws + WS_ATTNW);
  h16*   ddT   = (h16*)(ws + WS_DDT);
  h16*   ddB   = (h16*)(ws + WS_DDB);
  h16*   decW  = (h16*)(ws + WS_DECW);
  float* decB  = (float*)(ws + WS_DECB);
  h16*   fcT   = (h16*)(ws + WS_FCT);
  h16*   encS  = (h16*)(ws + WS_ENC);

  const int nx = B_ * L_ * F_;
  k_cvt_x<<<(nx + 255) / 256, 256, 0, stream>>>(x, xh, nx);
  k_encw<<<(512 * 192 + 255) / 256, 256, 0, stream>>>(eWih, eWhh, ebih, ebhh, encW, encB);
  k_decw<<<(512 * 384 + 255) / 256, 256, 0, stream>>>(dWih, dWhh, dbih, dbhh, decW, decB);
  k_attnw<<<(L_ * F_ * 192 + 255) / 256, 256, 0, stream>>>(attn_w, attnW);
  k_dd<<<(NOUT_ * H_ * H_ + 255) / 256, 256, 0, stream>>>(dd_w, ddT, ddB);
  k_fct<<<(NOUT_ * 64 * H_ + 255) / 256, 256, 0, stream>>>(fc_w, fcT);

  fused_darnn<<<NBLK, TPB, SMEM_BYTES, stream>>>(
      xh, encW, encB, attnW, attn_b, ddT, ddB, dd_b, dl_w, dl_b,
      decW, decB, fcT, fc_b, out_w, out_b, encS, (float*)d_out);
}
